// EuclideanCodebook_13391708029056
// MI455X (gfx1250) — compile-verified, matrix operand order checked by reading, not run on hardware
//
#include <hip/hip_runtime.h>
#include <hip/hip_bf16.h>

#define DECAY 0.99f
#define EPS 1e-5f
#define D 512
#define KCB 1024
#define DCHUNKS 16  // 512 / 32

typedef __bf16 bf16_t;
typedef __attribute__((ext_vector_type(16))) __bf16 v16bf;
typedef __attribute__((ext_vector_type(8)))  __bf16 v8bf;
typedef __attribute__((ext_vector_type(4)))  __bf16 v4bf;
typedef __attribute__((ext_vector_type(8)))  float  v8f;
typedef __attribute__((ext_vector_type(4)))  float  v4f;

// ---------------------------------------------------------------------------
// Kernel 1: embed fp32 -> bf16 copy, |e|^2 per row, zero counts / embed_sum.
// grid = 1024 (one block per codebook row), block = 128.
// ---------------------------------------------------------------------------
__global__ __launch_bounds__(128) void vq_prep_kernel(
    const float* __restrict__ embed, bf16_t* __restrict__ embedbf,
    float* __restrict__ e2, float* __restrict__ embed_sum,
    float* __restrict__ counts) {
  const int k = blockIdx.x;
  const int t = threadIdx.x;
  v4f v = *(const v4f*)(embed + (size_t)k * D + t * 4);
  v4bf bv;
  bv[0] = (__bf16)v[0]; bv[1] = (__bf16)v[1];
  bv[2] = (__bf16)v[2]; bv[3] = (__bf16)v[3];
  *(v4bf*)(embedbf + (size_t)k * D + t * 4) = bv;

  float s = v[0]*v[0] + v[1]*v[1] + v[2]*v[2] + v[3]*v[3];
  #pragma unroll
  for (int m = 16; m >= 1; m >>= 1) s += __shfl_xor(s, m, 32);
  __shared__ float ls[4];
  if ((t & 31) == 0) ls[t >> 5] = s;
  __syncthreads();
  if (t == 0) e2[k] = ls[0] + ls[1] + ls[2] + ls[3];

  v4f z = {0.f, 0.f, 0.f, 0.f};
  *(v4f*)(embed_sum + (size_t)k * D + t * 4) = z;
  if (t == 0) counts[k] = 0.f;
}

// ---------------------------------------------------------------------------
// Kernel 2: fused distance GEMM (bf16 WMMA, fp32 acc) + argmax over K.
// One wave handles 16 rows of x. block = 128 (4 waves), grid = N/64.
// dist(row,k) = 2 * dot(x_row, e_k) - |e_k|^2   (|x|^2 dropped: argmax-invariant)
// ---------------------------------------------------------------------------
__global__ __launch_bounds__(128) void vq_dist_argmax_kernel(
    const float* __restrict__ x, const bf16_t* __restrict__ embedbf,
    const float* __restrict__ e2, int* __restrict__ ind,
    float* __restrict__ embed_ind_out) {
  const int wave = threadIdx.x >> 5;
  const int lane = threadIdx.x & 31;
  const int hi   = lane >> 4;      // 0: K-sub {0..7,16..23}, 1: +8
  const int lm   = lane & 15;
  const int rowBase = (blockIdx.x * 4 + wave) * 16;

  // ---- Build A fragments: 16 rows x 512 depth, bf16, held in registers ----
  const float* xrow = x + (size_t)(rowBase + lm) * D;
  v16bf a[DCHUNKS];
  #pragma unroll
  for (int c = 0; c < DCHUNKS; ++c) {
    const int dbase = c * 32 + hi * 8;
    v4f x0 = *(const v4f*)(xrow + dbase);
    v4f x1 = *(const v4f*)(xrow + dbase + 4);
    v4f x2 = *(const v4f*)(xrow + dbase + 16);
    v4f x3 = *(const v4f*)(xrow + dbase + 20);
    v16bf av;
    #pragma unroll
    for (int j = 0; j < 4; ++j) {
      av[j]      = (__bf16)x0[j];
      av[4 + j]  = (__bf16)x1[j];
      av[8 + j]  = (__bf16)x2[j];
      av[12 + j] = (__bf16)x3[j];
    }
    a[c] = av;
  }

  float bestv[8];
  int   besti[8];
  #pragma unroll
  for (int r = 0; r < 8; ++r) { bestv[r] = -3.4e38f; besti[r] = 0; }

  // ---- K loop: 64 tiles of 16 codes, 16 WMMAs each (depth 512) ----
  for (int kt = 0; kt < KCB / 16; ++kt) {
    const int kb = kt * 16;
    const bf16_t* erow = embedbf + (size_t)(kb + lm) * D;
    v8f acc = {};
    #pragma unroll
    for (int c = 0; c < DCHUNKS; ++c) {
      const int dbase = c * 32 + hi * 8;
      v8bf blo = *(const v8bf*)(erow + dbase);        // depth dbase..+7
      v8bf bhi = *(const v8bf*)(erow + dbase + 16);   // depth dbase+16..+23
      v16bf bv = __builtin_shufflevector(blo, bhi,
                   0,1,2,3,4,5,6,7,8,9,10,11,12,13,14,15);
      acc = __builtin_amdgcn_wmma_f32_16x16x32_bf16(
          false, a[c], false, bv, (short)0, acc, false, false);
    }
    const float e2k = e2[kb + lm];   // column = lane%16, same for all 8 rows
    #pragma unroll
    for (int r = 0; r < 8; ++r) {
      float dist = 2.0f * acc[r] - e2k;
      if (dist > bestv[r]) { bestv[r] = dist; besti[r] = kb + lm; }
    }
  }

  // ---- argmax across the 16 lanes of each half (rows r and r+8) ----
  #pragma unroll
  for (int r = 0; r < 8; ++r) {
    float v = bestv[r];
    int   i = besti[r];
    #pragma unroll
    for (int m = 8; m >= 1; m >>= 1) {
      float ov = __shfl_xor(v, m, 32);
      int   oi = __shfl_xor(i, m, 32);
      if (ov > v || (ov == v && oi < i)) { v = ov; i = oi; }  // first-max tiebreak
    }
    if (lm == 0) {
      int row = rowBase + hi * 8 + r;
      ind[row] = i;
      embed_ind_out[row] = (float)i;
    }
  }
}

// ---------------------------------------------------------------------------
// Kernel 3: quantize gather + scatter-add (counts, embed_sum).
// grid = N (one block per row), block = 128.
// ---------------------------------------------------------------------------
__global__ __launch_bounds__(128) void vq_gather_scatter_kernel(
    const float* __restrict__ x, const float* __restrict__ embed,
    const int* __restrict__ ind, float* __restrict__ quantize,
    float* __restrict__ embed_sum, float* __restrict__ counts) {
  const int row = blockIdx.x;
  const int t = threadIdx.x;
  const int k = ind[row];                       // uniform per block -> scalar
  v4f ev = *(const v4f*)(embed + (size_t)k * D + t * 4);
  v4f xv = *(const v4f*)(x + (size_t)row * D + t * 4);
  *(v4f*)(quantize + (size_t)row * D + t * 4) = ev;
  float* es = embed_sum + (size_t)k * D + t * 4;
  atomicAdd(es + 0, xv[0]);
  atomicAdd(es + 1, xv[1]);
  atomicAdd(es + 2, xv[2]);
  atomicAdd(es + 3, xv[3]);
  if (t == 0) atomicAdd(&counts[k], 1.0f);
}

// ---------------------------------------------------------------------------
// Kernel 4: new_cluster_size EMA + total n. single block of 1024.
// ---------------------------------------------------------------------------
__global__ __launch_bounds__(1024) void vq_cluster_ema_kernel(
    const float* __restrict__ cluster_size, const float* __restrict__ counts,
    float* __restrict__ out_ncs, float* __restrict__ nptr) {
  const int k = threadIdx.x;
  float ncs = cluster_size[k] * DECAY + (1.0f - DECAY) * counts[k];
  out_ncs[k] = ncs;
  float s = ncs;
  #pragma unroll
  for (int m = 16; m >= 1; m >>= 1) s += __shfl_xor(s, m, 32);
  __shared__ float ls[32];
  if ((k & 31) == 0) ls[k >> 5] = s;
  __syncthreads();
  if (k < 32) {
    float w = ls[k];
    #pragma unroll
    for (int m = 16; m >= 1; m >>= 1) w += __shfl_xor(w, m, 32);
    if (k == 0) nptr[0] = w;
  }
}

// ---------------------------------------------------------------------------
// Kernel 5: new_embed_avg EMA + laplace-smoothed new_embed.
// grid = 1024 (row per block), block = 128.
// ---------------------------------------------------------------------------
__global__ __launch_bounds__(128) void vq_embed_ema_kernel(
    const float* __restrict__ embed_avg, const float* __restrict__ embed_sum,
    const float* __restrict__ out_ncs, const float* __restrict__ nptr,
    float* __restrict__ out_nea, float* __restrict__ out_ne) {
  const int k = blockIdx.x;
  const int t = threadIdx.x;
  const float n = nptr[0];
  const float ncs = out_ncs[k];
  const float cs = (ncs + EPS) / (n + (float)KCB * EPS) * n;
  const float inv = 1.0f / cs;
  const size_t off = (size_t)k * D + t * 4;
  v4f ea = *(const v4f*)(embed_avg + off);
  v4f es = *(const v4f*)(embed_sum + off);
  v4f nea, ne;
  #pragma unroll
  for (int j = 0; j < 4; ++j) {
    nea[j] = ea[j] * DECAY + (1.0f - DECAY) * es[j];
    ne[j]  = nea[j] * inv;
  }
  *(v4f*)(out_nea + off) = nea;
  *(v4f*)(out_ne + off)  = ne;
}

// ---------------------------------------------------------------------------
extern "C" void kernel_launch(void* const* d_in, const int* in_sizes, int n_in,
                              void* d_out, int out_size, void* d_ws, size_t ws_size,
                              hipStream_t stream) {
  const float* x            = (const float*)d_in[0];  // [N, 512]
  const float* embed        = (const float*)d_in[1];  // [1024, 512]
  const float* cluster_size = (const float*)d_in[2];  // [1024]
  const float* embed_avg    = (const float*)d_in[3];  // [1024, 512]

  const int N = in_sizes[0] / D;                      // 65536

  // Output layout (concatenated, return order)
  float* out      = (float*)d_out;
  float* q        = out;                              // N*512
  float* eind     = q + (size_t)N * D;                // N
  float* out_ncs  = eind + N;                         // 1024
  float* out_nea  = out_ncs + KCB;                    // 1024*512
  float* out_ne   = out_nea + (size_t)KCB * D;        // 1024*512

  // Workspace layout
  char* ws = (char*)d_ws;
  bf16_t* embedbf  = (bf16_t*)ws;                      // 1,048,576 B
  float* e2        = (float*)(ws + 1048576);           //     4,096 B
  float* counts    = (float*)(ws + 1052672);           //     4,096 B
  float* embed_sum = (float*)(ws + 1056768);           // 2,097,152 B
  int*   ind       = (int*)(ws + 3153920);             //   N*4 B
  float* nptr      = (float*)(ws + 3153920 + (size_t)N * 4);

  vq_prep_kernel<<<KCB, 128, 0, stream>>>(embed, embedbf, e2, embed_sum, counts);
  vq_dist_argmax_kernel<<<N / 64, 128, 0, stream>>>(x, embedbf, e2, ind, eind);
  vq_gather_scatter_kernel<<<N, 128, 0, stream>>>(x, embed, ind, q, embed_sum, counts);
  vq_cluster_ema_kernel<<<1, 1024, 0, stream>>>(cluster_size, counts, out_ncs, nptr);
  vq_embed_ema_kernel<<<KCB, 128, 0, stream>>>(embed_avg, embed_sum, out_ncs, nptr,
                                               out_nea, out_ne);
}
